// SelfAttention_49606872268908
// MI455X (gfx1250) — matrix operand order repaired and checked by reading, not compiled, not run on hardware
//
#include <hip/hip_runtime.h>
#include <hip/hip_bf16.h>

// Problem constants (match reference)
#define BB 4
#define CC 256
#define NN 4096   // H*W = 64*64
#define RD 32     // C/8
#define TJ 64     // j-block per flash-attention iteration

// Padded LDS row pitches (in halves) to avoid bank conflicts on b128 reads
#define KROW 40   // K tile row = 32 halves, padded to 40 (80B -> 20-bank stride)
#define VROW 72   // V tile row = 64 halves, padded to 72 (144B -> 36-bank stride)
#define PROW 72   // P tile row = 64 halves, padded to 72

typedef __attribute__((ext_vector_type(16))) _Float16 v16h;
typedef __attribute__((ext_vector_type(8)))  _Float16 h8;
typedef __attribute__((ext_vector_type(8)))  float    v8f;
typedef __attribute__((ext_vector_type(4)))  int      v4i;

#define AS1 __attribute__((address_space(1)))
#define AS3 __attribute__((address_space(3)))

// ---- explicit global-addrspace accessors (avoid flat_* fallback) ----
__device__ inline h8    gload8(const _Float16* p) { return *(const AS1 h8*)p; }
__device__ inline float gloadf(const float* p)    { return *(const AS1 float*)p; }
__device__ inline void  gstoref(float* p, float v){ *(AS1 float*)p = v; }
__device__ inline void  gstoreh(_Float16* p, _Float16 v){ *(AS1 _Float16*)p = v; }
__device__ inline void  gstore8(_Float16* p, h8 v){ *(AS1 h8*)p = v; }

__device__ inline v16h cat8(h8 a, h8 b) {
  return __builtin_shufflevector(a, b, 0,1,2,3,4,5,6,7,8,9,10,11,12,13,14,15);
}

__device__ inline v8f wmma_f16(v16h a, v16h b, v8f c) {
  // D = A(16x32 f16) * B(32x16 f16) + C(16x16 f32)
  return __builtin_amdgcn_wmma_f32_16x16x32_f16(
      /*neg_a=*/false, a, /*neg_b=*/false, b,
      /*c_mod=*/(short)0, c, /*reuse_a=*/false, /*reuse_b=*/false);
}

// ---- CDNA5 async global->LDS copy path (ASYNCcnt), with safe fallback ----
#if defined(__has_builtin)
#  if __has_builtin(__builtin_amdgcn_global_load_async_to_lds_b128) && \
      __has_builtin(__builtin_amdgcn_s_wait_asynccnt)
#    define USE_ASYNC_LDS 1
#  endif
#endif
#ifndef USE_ASYNC_LDS
#  define USE_ASYNC_LDS 0
#endif

__device__ inline void copy16_g2l(const _Float16* g, _Float16* l) {
#if USE_ASYNC_LDS
  __builtin_amdgcn_global_load_async_to_lds_b128(
      (AS1 v4i*)g, (AS3 v4i*)l, /*offset=*/0, /*cpol=*/0);
#else
  *(h8*)l = gload8(g);
#endif
}

__device__ inline void wait_stage() {
#if USE_ASYNC_LDS
  __builtin_amdgcn_s_wait_asynccnt(0);
#endif
  __syncthreads();
}

// Stage one j-block: K [2*TJ rows x first 16 halves] + V tile [CC x TJ] into LDS.
// 256 threads; K = 256 x 16B chunks (1 op/thread), V = 2048 chunks (8 ops).
__device__ inline void stage_tiles(const _Float16* Kg, const _Float16* Vg,
                                   _Float16* kb, _Float16* vb, int tid) {
  {
    const int j = tid >> 1, c2 = tid & 1;
    copy16_g2l(Kg + j * RD + c2 * 8, kb + j * KROW + c2 * 8);
  }
  #pragma unroll
  for (int r = 0; r < 8; ++r) {
    const int t = tid + r * 256;
    const int c = t >> 3, c2 = t & 7;
    copy16_g2l(Vg + (size_t)c * NN + c2 * 8, vb + c * VROW + c2 * 8);
  }
}

// ---------------------------------------------------------------------------
// Kernel 1: QKV projection (1x1 conv == channel GEMM), f32 in -> f16 staging.
//   Qh, Kh: [B, N, RD]  (token-major: fragment loads are contiguous)
//   Vh:     [B, C, N]   (channel-major: PV B-fragments contiguous in j)
// ---------------------------------------------------------------------------
__global__ __launch_bounds__(32) void proj_qkv_kernel(
    const float* __restrict__ x,
    const float* __restrict__ Wq, const float* __restrict__ bq,
    const float* __restrict__ Wk, const float* __restrict__ bk,
    const float* __restrict__ Wv, const float* __restrict__ bv,
    _Float16* __restrict__ Qh, _Float16* __restrict__ Kh,
    _Float16* __restrict__ Vh)
{
  const int lane = threadIdx.x & 31;
  const int ln   = lane & 15;
  const bool hi  = lane >= 16;
  const int n0   = blockIdx.x * 16;
  const int b    = blockIdx.y;
  const int grp  = blockIdx.z;

  const float* Wrow[4];
  const float* brow[4];
  if (grp == 0) {
    Wrow[0] = Wq;            brow[0] = bq;
    Wrow[1] = Wq + 16 * CC;  brow[1] = bq + 16;
    Wrow[2] = Wk;            brow[2] = bk;
    Wrow[3] = Wk + 16 * CC;  brow[3] = bk + 16;
  } else {
    const int rbase = (grp - 1) * 64;
    #pragma unroll
    for (int mt = 0; mt < 4; ++mt) {
      Wrow[mt] = Wv + (size_t)(rbase + mt * 16) * CC;
      brow[mt] = bv + rbase + mt * 16;
    }
  }

  v8f acc[4];
  #pragma unroll
  for (int mt = 0; mt < 4; ++mt) acc[mt] = v8f{};

  for (int kt = 0; kt < 8; ++kt) {
    // B fragment from x[b, k, n]: column n = n0+ln, K rows strided by N
    const int kbase = kt * 32 + (hi ? 8 : 0);
    const float* xp = x + (size_t)b * CC * NN + (size_t)kbase * NN + n0 + ln;
    v16h bf;
    #pragma unroll
    for (int t = 0; t < 16; ++t) bf[t] = (_Float16)gloadf(xp + (size_t)(t + (t >= 8 ? 8 : 0)) * NN);

    #pragma unroll
    for (int mt = 0; mt < 4; ++mt) {
      const float* wp = Wrow[mt] + (size_t)ln * CC + kt * 32 + (hi ? 8 : 0);
      v16h af;
      #pragma unroll
      for (int t = 0; t < 8; ++t) {
        af[t]     = (_Float16)gloadf(wp + t);
        af[t + 8] = (_Float16)gloadf(wp + t + 16);
      }
      acc[mt] = wmma_f16(af, bf, acc[mt]);
    }
  }

  // Bias + convert + store
  const int rsub = hi ? 8 : 0;
  #pragma unroll
  for (int mt = 0; mt < 4; ++mt) {
    h8 pk;
    #pragma unroll
    for (int g = 0; g < 8; ++g)
      pk[g] = (_Float16)(acc[mt][g] + gloadf(brow[mt] + rsub + g));

    if (grp == 0) {
      _Float16* dst = (mt < 2) ? Qh : Kh;
      const int mtl = mt & 1;
      dst += ((size_t)b * NN + n0 + ln) * RD + mtl * 16 + rsub;
      gstore8(dst, pk);                     // 16B contiguous store
    } else {
      const int c = (grp - 1) * 64 + mt * 16 + rsub;
      _Float16* dst = Vh + ((size_t)b * CC + c) * NN + n0 + ln;
      #pragma unroll
      for (int g = 0; g < 8; ++g) gstoreh(dst + (size_t)g * NN, pk[g]);
    }
  }
}

// ---------------------------------------------------------------------------
// Kernel 2: flash-attention. grid = (N/64, B), 256 threads = 8 waves.
// Double-buffered async K/V staging into LDS; per-wave P transpose buffer.
// LDS layout (halves): K[2][64*KROW] | V[2][256*VROW] | P[8][16*PROW]
// ---------------------------------------------------------------------------
#define SM_K0 0
#define SM_KSZ (64 * KROW)
#define SM_V0 (2 * SM_KSZ)
#define SM_VSZ (256 * VROW)
#define SM_P  (SM_V0 + 2 * SM_VSZ)
#define SM_TOTAL_HALVES (SM_P + 8 * 16 * PROW)

__global__ __launch_bounds__(256) void attn_kernel(
    const _Float16* __restrict__ Qh, const _Float16* __restrict__ Kh,
    const _Float16* __restrict__ Vh,
    const float* __restrict__ x, const float* __restrict__ gamma,
    float* __restrict__ out)
{
  extern __shared__ __align__(16) _Float16 smem[];

  const int tid  = threadIdx.x;
  const int lane = tid & 31;
  const int w    = tid >> 5;
  const int iw   = w & 3;
  const int ch   = w >> 2;
  const int b    = blockIdx.y;
  const int i0   = blockIdx.x * 64 + iw * 16;
  const int c0   = ch * 128;
  const int ln   = lane & 15;
  const bool hi  = lane >= 16;

  _Float16* pbuf = smem + SM_P + (size_t)w * (16 * PROW);

  const _Float16* Kg0 = Kh + (size_t)b * NN * RD;   // token-major [N, RD]
  const _Float16* Vg0 = Vh + (size_t)b * CC * NN;   // channel-major [C, N]

  // Q A-fragment (rows i0..i0+15, full K=RD=32), held for the whole loop
  v16h qa;
  {
    const _Float16* qp = Qh + ((size_t)b * NN + i0 + ln) * RD + (hi ? 8 : 0);
    qa = cat8(gload8(qp), gload8(qp + 16));
  }

  v8f oacc[8];
  #pragma unroll
  for (int ct = 0; ct < 8; ++ct) oacc[ct] = v8f{};
  float mrow[8], lrow[8];
  #pragma unroll
  for (int g = 0; g < 8; ++g) { mrow[g] = -__builtin_inff(); lrow[g] = 0.f; }

  // Preload first j-block
  stage_tiles(Kg0, Vg0, smem + SM_K0, smem + SM_V0, tid);
  wait_stage();

  const int NITER = NN / TJ;
  for (int jb = 0; jb < NITER; ++jb) {
    const int j0  = jb * TJ;
    const int cur = jb & 1;
    _Float16* kcur = smem + SM_K0 + cur * SM_KSZ;
    _Float16* vcur = smem + SM_V0 + cur * SM_VSZ;
    _Float16* knxt = smem + SM_K0 + (cur ^ 1) * SM_KSZ;
    _Float16* vnxt = smem + SM_V0 + (cur ^ 1) * SM_VSZ;

    // Kick off async staging of the next block into the other buffer
    if (jb + 1 < NITER)
      stage_tiles(Kg0 + (size_t)(j0 + TJ) * RD, Vg0 + j0 + TJ, knxt, vnxt, tid);

    // ---- S = Q^T K  (4 WMMAs -> 16 x 64 f32 tile), K frags from LDS ----
    v8f s[4];
    #pragma unroll
    for (int jt = 0; jt < 4; ++jt) {
      const _Float16* kp = kcur + (jt * 16 + ln) * KROW + (hi ? 8 : 0);
      v16h kb2 = cat8(*(const h8*)kp, *(const h8*)(kp + 16));
      v8f z = v8f{};
      s[jt] = wmma_f16(qa, kb2, z);
    }

    // ---- online softmax stats (row m lives in VGPR g, one 16-lane half) ----
    float scale[8];
    #pragma unroll
    for (int g = 0; g < 8; ++g) {
      float v = fmaxf(fmaxf(s[0][g], s[1][g]), fmaxf(s[2][g], s[3][g]));
      #pragma unroll
      for (int off = 1; off < 16; off <<= 1)
        v = fmaxf(v, __shfl_xor(v, off, 32));
      const float mnew = fmaxf(mrow[g], v);
      scale[g] = __expf(mrow[g] - mnew);
      mrow[g] = mnew;
    }

    #pragma unroll
    for (int jt = 0; jt < 4; ++jt)
      #pragma unroll
      for (int g = 0; g < 8; ++g)
        s[jt][g] = __expf(s[jt][g] - mrow[g]);

    #pragma unroll
    for (int g = 0; g < 8; ++g) {
      float v = s[0][g] + s[1][g] + s[2][g] + s[3][g];
      #pragma unroll
      for (int off = 1; off < 16; off <<= 1)
        v += __shfl_xor(v, off, 32);
      lrow[g] = lrow[g] * scale[g] + v;
    }
    #pragma unroll
    for (int ct = 0; ct < 8; ++ct)
      #pragma unroll
      for (int g = 0; g < 8; ++g) oacc[ct][g] *= scale[g];

    // ---- D-layout -> A-layout transpose of P through private LDS ----
    #pragma unroll
    for (int jt = 0; jt < 4; ++jt) {
      #pragma unroll
      for (int g = 0; g < 8; ++g) {
        const int m = g + (hi ? 8 : 0);
        pbuf[m * PROW + jt * 16 + ln] = (_Float16)s[jt][g];
      }
    }
    // (same-wave LDS producer/consumer: DScnt wait inserted by compiler)

    // ---- O += P * V^T  (2 K-steps x 8 c-subtiles = 16 WMMAs) ----
    #pragma unroll
    for (int jt2 = 0; jt2 < 2; ++jt2) {
      const _Float16* pp = pbuf + ln * PROW + jt2 * 32 + (hi ? 8 : 0);
      v16h pa = cat8(*(const h8*)pp, *(const h8*)(pp + 16));
      #pragma unroll
      for (int ct = 0; ct < 8; ++ct) {
        const _Float16* vp =
            vcur + (size_t)(c0 + ct * 16 + ln) * VROW + jt2 * 32 + (hi ? 8 : 0);
        v16h vb2 = cat8(*(const h8*)vp, *(const h8*)(vp + 16));
        oacc[ct] = wmma_f16(pa, vb2, oacc[ct]);
      }
    }

    // Drain this wave's async copies, then WG barrier before buffer swap
    wait_stage();
  }

  // ---- epilogue: O/l, gamma*O + x ----
  const float g0 = gloadf(gamma);
  #pragma unroll
  for (int ct = 0; ct < 8; ++ct) {
    const int c = c0 + ct * 16 + ln;
    #pragma unroll
    for (int g = 0; g < 8; ++g) {
      const int i = i0 + g + (hi ? 8 : 0);
      const size_t idx = ((size_t)b * CC + c) * NN + i;
      const float o = oacc[ct][g] / lrow[g];
      gstoref(out + idx, g0 * o + gloadf(x + idx));
    }
  }
}

// ---------------------------------------------------------------------------
extern "C" void kernel_launch(void* const* d_in, const int* in_sizes, int n_in,
                              void* d_out, int out_size, void* d_ws, size_t ws_size,
                              hipStream_t stream) {
  const float* x     = (const float*)d_in[0];
  const float* Wq    = (const float*)d_in[1];
  const float* bq    = (const float*)d_in[2];
  const float* Wk    = (const float*)d_in[3];
  const float* bk    = (const float*)d_in[4];
  const float* Wv    = (const float*)d_in[5];
  const float* bv    = (const float*)d_in[6];
  const float* gamma = (const float*)d_in[7];
  float* out = (float*)d_out;

  // f16 staging buffers in workspace (~10 MB total)
  _Float16* Qh = (_Float16*)d_ws;
  _Float16* Kh = Qh + (size_t)BB * NN * RD;
  _Float16* Vh = Kh + (size_t)BB * NN * RD;

  // Projection: 1 wave/block, grid (N/16, B, 5)
  dim3 pg(NN / 16, BB, 5);
  proj_qkv_kernel<<<pg, 32, 0, stream>>>(x, Wq, bq, Wk, bk, Wv, bv, Qh, Kh, Vh);

  // Flash attention: 8 waves/block, grid (N/64, B) = 256 WGs,
  // ~100 KB dynamic LDS (double-buffered K/V tiles + per-wave P buffers)
  dim3 ag(NN / 64, BB);
  const size_t smem_bytes = (size_t)SM_TOTAL_HALVES * sizeof(_Float16);
  attn_kernel<<<ag, 256, smem_bytes, stream>>>(Qh, Kh, Vh, x, gamma, out);
}
